// PoolingAggregator_1382979469562
// MI455X (gfx1250) — matrix operand order repaired and checked by reading, not compiled, not run on hardware
//
#include <hip/hip_runtime.h>
#include <cstdint>

// ---------------------------------------------------------------------------
// GraphSAGE pooling aggregator for MI455X (gfx1250, wave32, WMMA)
//   h      = relu(gather(features, neighbours) @ W1 + b1)   [640k x 128 GEMM]
//   pooled = mean_k(h)
//   out    = relu(concat(node_feat, pooled) @ neigh_weights)
// Main GEMM on v_wmma_f32_16x16x32_bf16; W1 staged transposed-bf16 in LDS so
// B-fragments are single contiguous 32B LDS loads (no per-frag converts).
// ---------------------------------------------------------------------------

typedef __attribute__((ext_vector_type(16))) __bf16 v16bf;
typedef __attribute__((ext_vector_type(2)))  __bf16 bf16x2;
typedef __attribute__((ext_vector_type(8)))  float  v8f;

// 128-bit int vector matching the async-LDS builtin's parameter type
typedef int v4i_vs __attribute__((vector_size(16)));
typedef __attribute__((address_space(1))) v4i_vs gv4i;   // global
typedef __attribute__((address_space(3))) v4i_vs lv4i;   // LDS

#define D_DIM        128
#define K_NB         32
#define OUT_DIM      128
#define NODES_PER_BLK 4
#define THREADS      256                      // 8 wave32

// Dynamic LDS layout (total 104448 B; 320KB/WGP -> 3 blocks per WGP)
#define SM_A_OFF    0          // f32  [128][128] 64KB  gathered neighbor tile
#define SM_W1T_OFF  65536      // bf16 [128][128] 32KB  W1 transposed: W1T[n][k]
#define SM_POOL_OFF 98304      // f32  [4][128]    2KB  per-node pooled sums
#define SM_CC_OFF   100352     // f32  [4][256]    4KB  concat(node_feat, pooled/K)
#define SM_BYTES    104448

// ---- async global->LDS copy (CDNA5 GLOBAL_LOAD_ASYNC_TO_LDS_B128), guarded --
#if defined(__has_builtin)
#if __has_builtin(__builtin_amdgcn_global_load_async_to_lds_b128)
#define HAVE_ASYNC_LDS 1
#endif
#endif

__device__ __forceinline__ void copy16_to_lds(const float* g, float* l) {
#ifdef HAVE_ASYNC_LDS
  __builtin_amdgcn_global_load_async_to_lds_b128((gv4i*)g, (lv4i*)l, 0, 0);
#else
  *(float4*)l = *(const float4*)g;
#endif
}

__device__ __forceinline__ void wait_async_copies() {
#ifdef HAVE_ASYNC_LDS
#if __has_builtin(__builtin_amdgcn_s_wait_asynccnt)
  __builtin_amdgcn_s_wait_asynccnt(0);
#else
  asm volatile("s_wait_asynccnt 0" ::: "memory");
#endif
#endif
}

__global__ __launch_bounds__(THREADS)
void sage_pool_kernel(const float* __restrict__ features,
                      const long long* __restrict__ node,
                      const long long* __restrict__ neighbours,
                      const float* __restrict__ W1,
                      const float* __restrict__ b1,
                      const float* __restrict__ nw,     // (2D, OUT) row-major
                      float* __restrict__ out)
{
  extern __shared__ char smem[];
  float*  sA    = (float*) (smem + SM_A_OFF);    // [128][128] f32
  __bf16* sW1T  = (__bf16*)(smem + SM_W1T_OFF);  // [n=128][k=128] bf16
  float*  sPool = (float*) (smem + SM_POOL_OFF); // [4][128]
  float*  sCC   = (float*) (smem + SM_CC_OFF);   // [4][256]

  const int tid  = threadIdx.x;
  const int wave = tid >> 5;        // 0..7 (wave32)
  const int lane = tid & 31;
  const int lm   = lane & 15;
  const int hi   = lane >> 4;       // 0/1: half-row group of the frag
  const int node0 = blockIdx.x * NODES_PER_BLK;

  // L2 warm-up hint for the (L2-resident) weight matrices
  __builtin_prefetch(nw + (tid << 6), 0, 1);
  __builtin_prefetch(W1 + (tid << 6), 0, 1);

  // zero the pooled accumulator (512 floats, 256 threads)
  sPool[tid]       = 0.0f;
  sPool[tid + 256] = 0.0f;

  // ---- Phase 1a: async-DMA the gathered neighbor rows (f32) into LDS ------
  // One feature row = 128 f32 = 512B = 32 lanes x 16B -> one async op per row.
#pragma unroll 4
  for (int i = 0; i < 16; ++i) {
    const int r  = wave * 16 + i;                      // 0..127
    const int nl = r >> 5;                             // node-local 0..3
    const long long nb = neighbours[(long long)(node0 + nl) * K_NB + (r & 31)];
    copy16_to_lds(features + nb * D_DIM + lane * 4, sA + r * D_DIM + lane * 4);
  }
  // node features straight into the concat buffer (first 128 of each row)
  if (wave < NODES_PER_BLK) {
    const long long nd = node[node0 + wave];
    copy16_to_lds(features + nd * D_DIM + lane * 4, sCC + wave * 256 + lane * 4);
  }

  // ---- Phase 1b (overlaps the DMA): stage W1 transposed as bf16 ------------
  // W1T[n][k] = bf16(W1[k][n]); each thread owns one (n, k-half) strip.
  {
    const int n  = tid >> 1;            // 0..127
    const int kh = (tid & 1) * 64;      // k range base
#pragma unroll 8
    for (int kk = 0; kk < 64; kk += 2) {
      const float a = W1[(kh + kk)     * D_DIM + n];
      const float b = W1[(kh + kk + 1) * D_DIM + n];
      bf16x2 p = {(__bf16)a, (__bf16)b};
      *(bf16x2*)(sW1T + n * D_DIM + kh + kk) = p;   // 4B-aligned ds_store_b32
    }
  }
  wait_async_copies();
  __syncthreads();

  // ---- Phase 2: 128x128x128 GEMM on v_wmma_f32_16x16x32_bf16 ---------------
  // Wave w owns M-tile w (rows 16w..16w+15). 8 N-tiles x 4 K-steps = 32 WMMA.
  // A frag (16x32 bf16): lane row lm; element j -> K=abase+j,
  // abase = hi?16:0. Built once, reused for 8 N-tiles.
  const int arow  = wave * 16 + lm;
  const int abase = hi ? 16 : 0;
  v16bf afr[4];
#pragma unroll
  for (int ks = 0; ks < 4; ++ks) {
    const float* ap = sA + arow * D_DIM + ks * 32 + abase;
#pragma unroll
    for (int j = 0; j < 8; ++j) {
      afr[ks][j]     = (__bf16)ap[j];
      afr[ks][8 + j] = (__bf16)ap[8 + j];
    }
  }

  const int nodeL = wave >> 1;          // this wave's 16 rows belong to node w/2
  const int kb    = hi ? 16 : 0;        // B frag: lanes 0-15 K=0..15, 16-31 K=16..31
#pragma unroll
  for (int nt = 0; nt < 8; ++nt) {
    const int n = nt * 16 + lm;
    const float bias = b1[n];
    v8f c;
#pragma unroll
    for (int v = 0; v < 8; ++v) c[v] = bias;

#pragma unroll
    for (int ks = 0; ks < 4; ++ks) {
      // 16 contiguous bf16 of W1T row n -> one 32B LDS load (2x ds_load_b128)
      const v16bf bfr = *(const v16bf*)(sW1T + n * D_DIM + ks * 32 + kb);
      c = __builtin_amdgcn_wmma_f32_16x16x32_bf16(
              false, afr[ks], false, bfr, (short)0, c, false, false);
    }
    // relu + partial column sum of this wave's 16 rows, atomically pooled
    float s = 0.0f;
#pragma unroll
    for (int v = 0; v < 8; ++v) s += fmaxf(c[v], 0.0f);
    atomicAdd(&sPool[nodeL * D_DIM + n], s);   // ds_add_f32
  }
  __syncthreads();

  // ---- Phase 3: pooled/K into second half of concat buffer ----------------
#pragma unroll
  for (int rep = 0; rep < 2; ++rep) {
    const int idx = tid + rep * 256;           // 0..511
    const int nd  = idx >> 7;
    const int dc  = idx & 127;
    sCC[nd * 256 + 128 + dc] = sPool[nd * D_DIM + dc] * (1.0f / (float)K_NB);
  }
  __syncthreads();

  // ---- Phase 4: tiny (4x256)@(256x128) GEMM + relu (VALU; ~6% of FLOPs) --
#pragma unroll
  for (int rep = 0; rep < 2; ++rep) {
    const int o  = tid + rep * 256;            // 0..511
    const int nd = o >> 7;
    const int e  = o & 127;
    const float* cc = sCC + nd * 256;
    float acc = 0.0f;
#pragma unroll 8
    for (int d = 0; d < 256; ++d)
      acc = fmaf(cc[d], nw[d * OUT_DIM + e], acc);
    out[(long long)(node0 + nd) * OUT_DIM + e] = fmaxf(acc, 0.0f);
  }
}

extern "C" void kernel_launch(void* const* d_in, const int* in_sizes, int n_in,
                              void* d_out, int out_size, void* d_ws, size_t ws_size,
                              hipStream_t stream) {
  (void)n_in; (void)d_ws; (void)ws_size; (void)out_size;
  const float*     features   = (const float*)d_in[0];
  const long long* node       = (const long long*)d_in[1];   // int64 (B,1)
  const long long* neighbours = (const long long*)d_in[2];   // int64 (B,K)
  const float*     W1         = (const float*)d_in[3];
  const float*     b1         = (const float*)d_in[4];
  const float*     nw         = (const float*)d_in[5];
  float*           out        = (float*)d_out;

  const int batch  = in_sizes[1];              // node has B elements
  const int blocks = batch / NODES_PER_BLK;    // B=20000 -> 5000 blocks

  // allow >64KB dynamic LDS (CDNA5: up to 320KB per workgroup)
  (void)hipFuncSetAttribute((const void*)sage_pool_kernel,
                            hipFuncAttributeMaxDynamicSharedMemorySize, SM_BYTES);

  sage_pool_kernel<<<blocks, THREADS, SM_BYTES, stream>>>(
      features, node, neighbours, W1, b1, nw, out);
}
